// WordRNN_17274358465265
// MI455X (gfx1250) — matrix operand order repaired and checked by reading, not compiled, not run on hardware
//
#include <hip/hip_runtime.h>

// ---------------- problem constants (from reference) ----------------
#define EMBED  50
#define SEQ    100
#define NLABEL 15
#define HID    20
#define G3     60          // 3*HID
#define BATCH  8192
#define EPAD   64          // padded embedding width (f16 table), col 50 = 1.0 bias feature

#define WPB    2           // waves per block
#define BLOCK  (32 * WPB)  // wave32

typedef __attribute__((ext_vector_type(16))) _Float16 v16h;
typedef __attribute__((ext_vector_type(8)))  _Float16 v8h;
typedef __attribute__((ext_vector_type(8)))  float    v8f;

__device__ __forceinline__ v8f wmma_f16(v16h a, v16h b, v8f c) {
    return __builtin_amdgcn_wmma_f32_16x16x32_f16(
        false, a, false, b, (short)0, c, false, false);
}

__device__ __forceinline__ float sigmoidf_(float x) {
    return __frcp_rn(1.0f + __expf(-x));
}
// tanh(u) = 2*sigmoid(2u) - 1 : one hw transcendental instead of libm tanhf
__device__ __forceinline__ float tanhf_fast(float x) {
    return __builtin_fmaf(2.0f, __frcp_rn(1.0f + __expf(-2.0f * x)), -1.0f);
}

// B fragment: this lane's column, K chunks p[0..7] and p[16..23] (two aligned 16B chunks)
__device__ __forceinline__ v16h load_bfrag(const _Float16* p) {
    const v8h* v = (const v8h*)p;
    v8h lo = v[0], hi = v[2];
    v16h r;
#pragma unroll
    for (int e = 0; e < 8; ++e) { r[e] = lo[e]; r[8 + e] = hi[e]; }
    return r;
}

// A fragment from a 32-wide f16 row: chunks at k=kb..kb+7 and k=kb+16..kb+23
__device__ __forceinline__ v16h load_afrag_row32(const _Float16* row, int hi) {
    const v8h* v = (const v8h*)row;
    v8h c0 = v[hi], c1 = v[hi + 2];
    v16h r;
#pragma unroll
    for (int e = 0; e < 8; ++e) { r[e] = c0[e]; r[8 + e] = c1[e]; }
    return r;
}

// ---------- pre-pass: f32 embedding table -> padded f16 table with bias column ----------
__global__ __launch_bounds__(256) void emb_to_f16_kernel(
    const float* __restrict__ emb, _Float16* __restrict__ et, int vocab)
{
    long i = (long)blockIdx.x * blockDim.x + threadIdx.x;   // over vocab * EPAD
    if (i >= (long)vocab * EPAD) return;
    int v = (int)(i >> 6), k = (int)(i & (EPAD - 1));
    float val = (k < EMBED) ? emb[(long)v * EMBED + k] : (k == EMBED ? 1.0f : 0.0f);
    et[i] = (_Float16)val;
}

// ---------------------------- fused scan kernel ----------------------------
template <bool F16TAB>
__global__ __launch_bounds__(BLOCK) void wordrnn_fused_kernel(
    const int*      __restrict__ x,     // (BATCH, SEQ)
    const float*    __restrict__ emb,   // (VOCAB, EMBED)  (used when !F16TAB)
    const _Float16* __restrict__ et,    // (VOCAB, EPAD)   (used when F16TAB)
    const float* __restrict__ W1, const float* __restrict__ U1, const float* __restrict__ b1,
    const float* __restrict__ W2, const float* __restrict__ U2, const float* __restrict__ b2,
    const float* __restrict__ Wd, const float* __restrict__ bd,
    float*       __restrict__ out)      // (BATCH, NLABEL)
{
    // f16 weights, transposed (N-major), zero padded; bias folded in at k==HID / k==EMBED
    __shared__ __align__(16) _Float16 WT1[64][64];
    __shared__ __align__(16) _Float16 UT1[64][32];
    __shared__ __align__(16) _Float16 WT2[64][32];
    __shared__ __align__(16) _Float16 UT2[64][32];
    __shared__ __align__(16) _Float16 WdT[16][32];
    // per-wave state: f32 (recurrence precision) + f16 tile (WMMA A operand, 1.0 at k==HID)
    __shared__ __align__(16) float    h1f[WPB][16][32];
    __shared__ __align__(16) float    h2f[WPB][16][32];
    __shared__ __align__(16) _Float16 h1h[WPB][16][32];
    __shared__ __align__(16) _Float16 h2h[WPB][16][32];
    __shared__ __align__(16) float    mxs[WPB][16][64];
    __shared__ __align__(16) float    mhs[WPB][16][64];

    const int tid = threadIdx.x;

    // ---------------- prologue: stage weights (+bias rows) ----------------
    for (int i = tid; i < 64 * 64; i += BLOCK) {
        int n = i >> 6, k = i & 63;
        float v = 0.0f;
        if (n < G3) { if (k < EMBED) v = W1[k * G3 + n]; else if (k == EMBED) v = b1[n]; }
        WT1[n][k] = (_Float16)v;
    }
    for (int i = tid; i < 64 * 32; i += BLOCK) {
        int n = i >> 5, k = i & 31;
        float vu1 = 0.0f, vw2 = 0.0f, vu2 = 0.0f;
        if (n < G3) {
            if (k < HID)       { vu1 = U1[k * G3 + n]; vw2 = W2[k * G3 + n]; vu2 = U2[k * G3 + n]; }
            else if (k == HID) { vu1 = b1[G3 + n];     vw2 = b2[n];          vu2 = b2[G3 + n]; }
        }
        UT1[n][k] = (_Float16)vu1; WT2[n][k] = (_Float16)vw2; UT2[n][k] = (_Float16)vu2;
    }
    for (int i = tid; i < 16 * 32; i += BLOCK) {
        int n = i >> 5, k = i & 31;
        float v = 0.0f;
        if (n < NLABEL) { if (k < HID) v = Wd[k * NLABEL + n]; else if (k == HID) v = bd[n]; }
        WdT[n][k] = (_Float16)v;
    }

    const int w    = tid >> 5;
    const int lane = tid & 31;
    for (int i = lane; i < 16 * 32; i += 32) {
        int k = i & 31;
        (&h1f[w][0][0])[i] = 0.0f;
        (&h2f[w][0][0])[i] = 0.0f;
        _Float16 one = (_Float16)(k == HID ? 1.0f : 0.0f);   // bias feature
        (&h1h[w][0][0])[i] = one;
        (&h2h[w][0][0])[i] = one;
    }
    __syncthreads();

    // WMMA lane decomposition (wave32 layouts, CDNA5 ISA 7.12.2)
    const int m  = lane & 15;        // A row / B,C,D column for this lane
    const int hw = lane >> 4;        // half-wave select
    const int kb = hw * 8;           // A-frag K base
    const int bk = hw * 8;           // B-frag K base
    const long b0 = (long)blockIdx.x * (WPB * 16) + (long)w * 16;
    const int* xrow = x + (b0 + m) * SEQ;

    for (int t = 0; t < SEQ; ++t) {
        const int idx = xrow[t];
        v16h ax0, ax1;
        if (F16TAB) {
            // padded f16 table: 4 aligned 16B gathers, bias column baked in
            const v8h* er = (const v8h*)(et + (long)idx * EPAD);
            v8h c0 = er[hw], c1 = er[hw + 2], c2 = er[hw + 4], c3 = er[hw + 6];
#pragma unroll
            for (int e = 0; e < 8; ++e) {
                ax0[e] = c0[e]; ax0[8 + e] = c1[e];
                ax1[e] = c2[e]; ax1[8 + e] = c3[e];
            }
        } else {
            const float* er = emb + (long)idx * EMBED;
#pragma unroll
            for (int e = 0; e < 16; ++e) {
                int k0 = kb + (e < 8 ? e : e + 8);   // 0..31, always < EMBED
                int k1 = 32 + k0;                    // 32..63
                ax0[e] = (_Float16)er[k0];
                ax1[e] = (_Float16)(k1 < EMBED ? er[k1] : (k1 == EMBED ? 1.0f : 0.0f));
            }
        }
        v16h ah1 = load_afrag_row32(&h1h[w][m][0], hw);

        // -------- GRU1: mh = [h1,1]@[U1;br1] ; mx = [x,1]@[W1;bi1]  (C = 0) --------
#pragma unroll
        for (int nt = 0; nt < 4; ++nt) {
            const int col = nt * 16 + m;
            v8f ch = {};
            ch = wmma_f16(ah1, load_bfrag(&UT1[col][bk]), ch);
#pragma unroll
            for (int r = 0; r < 8; ++r) mhs[w][r + hw * 8][col] = ch[r];
            v8f cx = {};
            cx = wmma_f16(ax0, load_bfrag(&WT1[col][bk]),      cx);
            cx = wmma_f16(ax1, load_bfrag(&WT1[col][32 + bk]), cx);
#pragma unroll
            for (int r = 0; r < 8; ++r) mxs[w][r + hw * 8][col] = cx[r];
        }
        __syncthreads();

        // -------- GRU1 gates: 16x20 elements, 10 per lane --------
#pragma unroll
        for (int q = 0; q < 10; ++q) {
            const int p  = lane + 32 * q;
            const int mm = p / HID;
            const int j  = p % HID;
            float z  = sigmoidf_ (mxs[w][mm][j]           + mhs[w][mm][j]);
            float r  = sigmoidf_ (mxs[w][mm][j + HID]     + mhs[w][mm][j + HID]);
            float hh = tanhf_fast(mxs[w][mm][j + 2 * HID] + r * mhs[w][mm][j + 2 * HID]);
            float hn = z * h1f[w][mm][j] + (1.0f - z) * hh;
            h1f[w][mm][j] = hn;
            h1h[w][mm][j] = (_Float16)hn;
        }
        __syncthreads();

        // -------- GRU2: input = fresh h1 (K=HID, one K tile) --------
        v16h ax2 = load_afrag_row32(&h1h[w][m][0], hw);
        v16h ah2 = load_afrag_row32(&h2h[w][m][0], hw);
#pragma unroll
        for (int nt = 0; nt < 4; ++nt) {
            const int col = nt * 16 + m;
            v8f ch = {};
            ch = wmma_f16(ah2, load_bfrag(&UT2[col][bk]), ch);
#pragma unroll
            for (int r = 0; r < 8; ++r) mhs[w][r + hw * 8][col] = ch[r];
            v8f cx = {};
            cx = wmma_f16(ax2, load_bfrag(&WT2[col][bk]), cx);
#pragma unroll
            for (int r = 0; r < 8; ++r) mxs[w][r + hw * 8][col] = cx[r];
        }
        __syncthreads();

        // -------- GRU2 gates --------
#pragma unroll
        for (int q = 0; q < 10; ++q) {
            const int p  = lane + 32 * q;
            const int mm = p / HID;
            const int j  = p % HID;
            float z  = sigmoidf_ (mxs[w][mm][j]           + mhs[w][mm][j]);
            float r  = sigmoidf_ (mxs[w][mm][j + HID]     + mhs[w][mm][j + HID]);
            float hh = tanhf_fast(mxs[w][mm][j + 2 * HID] + r * mhs[w][mm][j + 2 * HID]);
            float hn = z * h2f[w][mm][j] + (1.0f - z) * hh;
            h2f[w][mm][j] = hn;
            h2h[w][mm][j] = (_Float16)hn;
        }
        __syncthreads();
    }

    // ---------------- dense head: logits = [h2,1] @ [Wd;bd]  (one WMMA) ----------------
    v16h ah2 = load_afrag_row32(&h2h[w][m][0], hw);
    v8f c = {};
    c = wmma_f16(ah2, load_bfrag(&WdT[m][bk]), c);
    if (m < NLABEL) {
#pragma unroll
        for (int r = 0; r < 8; ++r) {
            const long row = b0 + (r + hw * 8);
            out[row * NLABEL + m] = c[r];
        }
    }
}

extern "C" void kernel_launch(void* const* d_in, const int* in_sizes, int n_in,
                              void* d_out, int out_size, void* d_ws, size_t ws_size,
                              hipStream_t stream) {
    (void)n_in; (void)out_size;
    const int*   x    = (const int*)  d_in[0];
    const float* emb  = (const float*)d_in[1];
    const float* W1   = (const float*)d_in[2];
    const float* U1   = (const float*)d_in[3];
    const float* b1   = (const float*)d_in[4];
    const float* W2   = (const float*)d_in[5];
    const float* U2   = (const float*)d_in[6];
    const float* b2   = (const float*)d_in[7];
    const float* Wd   = (const float*)d_in[8];
    const float* bd   = (const float*)d_in[9];
    float* out = (float*)d_out;

    const int  vocab   = in_sizes[1] / EMBED;
    const size_t etab  = (size_t)vocab * EPAD * sizeof(_Float16);
    const int  grid    = BATCH / (WPB * 16);

    if (ws_size >= etab) {
        _Float16* et = (_Float16*)d_ws;
        const long n = (long)vocab * EPAD;
        emb_to_f16_kernel<<<(int)((n + 255) / 256), 256, 0, stream>>>(emb, et, vocab);
        wordrnn_fused_kernel<true><<<grid, BLOCK, 0, stream>>>(
            x, emb, et, W1, U1, b1, W2, U2, b2, Wd, bd, out);
    } else {
        wordrnn_fused_kernel<false><<<grid, BLOCK, 0, stream>>>(
            x, emb, nullptr, W1, U1, b1, W2, U2, b2, Wd, bd, out);
    }
}
